// xERTE_20598663152053
// MI455X (gfx1250) — matrix operand order repaired and checked by reading, not compiled, not run on hardware
//
#include <hip/hip_runtime.h>

typedef __attribute__((ext_vector_type(16))) _Float16 v16h;
typedef __attribute__((ext_vector_type(8)))  _Float16 v8h;
typedef __attribute__((ext_vector_type(8)))  float    v8f;

#define NSEG  100000
#define MSTR  264   // LDS row stride (halves) for M: 132 dwords -> 4-bank rotation, conflict-free
#define RSTR  264   // LDS row stride for right-tile rows (A operand)
#define LSTR  24    // LDS row stride for transposed left-tile (12 dwords -> conflict-free b128 reads)
#define NWAVE 8

// monotonic encoding of float -> unsigned for atomicMax-based segment max
__device__ __forceinline__ unsigned encF(float f) {
  unsigned u = __float_as_uint(f);
  return (u & 0x80000000u) ? ~u : (u | 0x80000000u);
}
__device__ __forceinline__ float decF(unsigned e) {
  unsigned u = (e & 0x80000000u) ? (e & 0x7fffffffu) : ~e;
  return __uint_as_float(u);
}

__device__ __forceinline__ void st2(_Float16* dst, float x, float y) {
  union { _Float16 h[2]; unsigned u; } p;
  p.h[0] = (_Float16)x; p.h[1] = (_Float16)y;
  *(unsigned*)dst = p.u;   // 4B-aligned packed half2 store
}

__global__ void k_init_seg(unsigned* __restrict__ smax, float* __restrict__ den, int nseg) {
  int i = blockIdx.x * blockDim.x + threadIdx.x;
  if (i < nseg) { smax[i] = 0u; den[i] = 0.0f; }
}

// M = Wq^T @ Wk  (256x256 fp32 accumulate), stored row-major f16
__global__ void k_bilinear_M(const float* __restrict__ Wq, const float* __restrict__ Wk,
                             _Float16* __restrict__ Mh) {
  int c = blockIdx.x;    // row of M
  int d = threadIdx.x;   // col of M
  float acc = 0.0f;
  #pragma unroll 8
  for (int n = 0; n < 256; ++n)
    acc = fmaf(Wq[n * 256 + c], Wk[n * 256 + d], acc);
  Mh[c * 256 + d] = (_Float16)acc;
}

__global__ void __launch_bounds__(256, 1)
k_logits_wmma(const float* __restrict__ memb,
              const float* __restrict__ rel,
              const float* __restrict__ qsrc,
              const float* __restrict__ qrel,
              const _Float16* __restrict__ Mh,
              const int* __restrict__ src_idx,
              const int* __restrict__ dst_idx,
              const int* __restrict__ query_idx,
              const int* __restrict__ seg_ids,
              float* __restrict__ logits,
              unsigned* __restrict__ smax,
              int E, int nTiles) {
  __shared__ _Float16 sM[256 * MSTR];          // 135168 B : bilinear matrix (block-shared)
  __shared__ _Float16 sR[NWAVE * 16 * RSTR];   //  67584 B : per-wave right rows [16 edges][256]
  __shared__ _Float16 sL[NWAVE * 256 * LSTR];  //  98304 B : per-wave left, transposed [256][16]

  const int tid  = threadIdx.x;
  const int lane = tid & 31;
  const int wv   = tid >> 5;
  const int l15  = lane & 15;
  const int g    = (lane >> 4) & 1;

  // cooperative load of f16 M into padded LDS (32 iters of b128 per thread)
  for (int i = tid; i < 256 * 32; i += 256) {
    int row = i >> 5, ch = i & 31;
    *(v8h*)&sM[row * MSTR + ch * 8] = *(const v8h*)(Mh + row * 256 + ch * 8);
  }
  __syncthreads();

  const int  tile   = blockIdx.x * NWAVE + wv;
  const bool active = tile < nTiles;
  const int  tb     = active ? tile * 16 : 0;

  _Float16* rt = &sR[wv * 16 * RSTR];
  _Float16* lt = &sL[wv * 256 * LSTR];

  // ---- gather + stage 16 edges (rows are 256B contiguous; 2 floats per lane) ----
  for (int e = 0; e < 16; ++e) {
    int edge = tb + e; if (edge >= E) edge = E - 1;
    int si = src_idx[edge], di = dst_idx[edge], qi = query_idx[edge];  // wave-uniform -> s_loads
    const int dbase = lane * 2;
    const float2 hv = *(const float2*)(memb + (size_t)si   * 64 + dbase);
    const float2 hj = *(const float2*)(memb + (size_t)di   * 64 + dbase);
    const float2 rl = *(const float2*)(rel  + (size_t)edge * 64 + dbase);
    const float2 qs = *(const float2*)(qsrc + (size_t)qi   * 64 + dbase);
    const float2 qr = *(const float2*)(qrel + (size_t)qi   * 64 + dbase);
    // right = [h_vj, rel, q_src, q_rel], row-major (A-operand rows)
    st2(&rt[e * RSTR +   0 + dbase], hj.x, hj.y);
    st2(&rt[e * RSTR +  64 + dbase], rl.x, rl.y);
    st2(&rt[e * RSTR + 128 + dbase], qs.x, qs.y);
    st2(&rt[e * RSTR + 192 + dbase], qr.x, qr.y);
    // left  = [h_vi, rel, q_src, q_rel], stored transposed: lt[dim][edge]
    lt[(  0 + dbase) * LSTR + e] = (_Float16)hv.x;
    lt[(  1 + dbase) * LSTR + e] = (_Float16)hv.y;
    lt[( 64 + dbase) * LSTR + e] = (_Float16)rl.x;
    lt[( 65 + dbase) * LSTR + e] = (_Float16)rl.y;
    lt[(128 + dbase) * LSTR + e] = (_Float16)qs.x;
    lt[(129 + dbase) * LSTR + e] = (_Float16)qs.y;
    lt[(192 + dbase) * LSTR + e] = (_Float16)qr.x;
    lt[(193 + dbase) * LSTR + e] = (_Float16)qr.y;
  }

  // ---- T = right @ M^T via WMMA, fused with P += T (.) left  ----
  const _Float16* arow = &rt[l15 * RSTR];  // A: lane row = edge l15; K-offset 8*g (+16)
  float P[8] = {0.f,0.f,0.f,0.f,0.f,0.f,0.f,0.f};

  for (int nt = 0; nt < 16; ++nt) {
    const _Float16* brow = &sM[(nt * 16 + l15) * MSTR];  // B: lane col n holds M row (nt*16+n)
    v8f acc = {0.f,0.f,0.f,0.f,0.f,0.f,0.f,0.f};
    #pragma unroll
    for (int kt = 0; kt < 8; ++kt) {
      const int kb = kt * 32;
      v8h a0 = *(const v8h*)(arow + kb + 8 * g);        // A K = kb+8g .. +7
      v8h a1 = *(const v8h*)(arow + kb + 8 * g + 16);   // A K = kb+8g+16 .. +23
      v8h b0 = *(const v8h*)(brow + kb + 8 * g);        // B K = kb+8g .. +7
      v8h b1 = *(const v8h*)(brow + kb + 8 * g + 16);   // B K = kb+8g+16 .. +23
      v16h A = __builtin_shufflevector(a0, a1, 0,1,2,3,4,5,6,7,8,9,10,11,12,13,14,15);
      v16h B = __builtin_shufflevector(b0, b1, 0,1,2,3,4,5,6,7,8,9,10,11,12,13,14,15);
      acc = __builtin_amdgcn_wmma_f32_16x16x32_f16(false, A, false, B, (short)0, acc,
                                                   false, false);
    }
    // C layout: lane n = nt*16+l15, VGPR r -> edge m = r + 8*g.  Read L[m][n] (8 halves, b128)
    v8h lf = *(const v8h*)&lt[(nt * 16 + l15) * LSTR + 8 * g];
    #pragma unroll
    for (int r = 0; r < 8; ++r) P[r] = fmaf((float)lf[r], acc[r], P[r]);
  }

  // rowsum across N: reduce over the 16 lanes of each half-group
  #pragma unroll
  for (int off = 1; off < 16; off <<= 1) {
    #pragma unroll
    for (int r = 0; r < 8; ++r) P[r] += __shfl_xor(P[r], off, 32);
  }

  if (active && l15 == 0) {
    #pragma unroll
    for (int r = 0; r < 8; ++r) {
      int edge = tb + g * 8 + r;                 // lane0 -> edges 0..7, lane16 -> 8..15
      if (edge < E) {
        logits[edge] = P[r];
        atomicMax(&smax[seg_ids[edge]], encF(P[r]));
      }
    }
  }
}

__global__ void k_seg_exp(const float* __restrict__ logits, const int* __restrict__ seg,
                          const unsigned* __restrict__ smax, float* __restrict__ den,
                          float* __restrict__ out, int E) {
  int i = blockIdx.x * blockDim.x + threadIdx.x;
  if (i >= E) return;
  int s = seg[i];
  float e = expf(logits[i] - decF(smax[s]));
  out[i] = e;
  atomicAdd(&den[s], e);
}

__global__ void k_seg_div(const int* __restrict__ seg, const float* __restrict__ den,
                          float* __restrict__ out, int E) {
  int i = blockIdx.x * blockDim.x + threadIdx.x;
  if (i >= E) return;
  out[i] = out[i] / den[seg[i]];
}

extern "C" void kernel_launch(void* const* d_in, const int* in_sizes, int n_in,
                              void* d_out, int out_size, void* d_ws, size_t ws_size,
                              hipStream_t stream) {
  const float* memb = (const float*)d_in[0];
  const float* rel  = (const float*)d_in[1];
  const float* qsrc = (const float*)d_in[2];
  const float* qrel = (const float*)d_in[3];
  const float* Wq   = (const float*)d_in[4];
  const float* Wk   = (const float*)d_in[5];
  const int* src_idx   = (const int*)d_in[6];
  const int* dst_idx   = (const int*)d_in[7];
  const int* query_idx = (const int*)d_in[8];
  const int* seg_ids   = (const int*)d_in[9];
  float* out = (float*)d_out;
  const int E = in_sizes[6];

  // workspace carve-up (~2.94 MB): Mh f16 | logits f32[E] | segmax u32 | den f32
  char* w = (char*)d_ws;
  _Float16* Mh = (_Float16*)w;  w += (((size_t)256 * 256 * 2) + 255) & ~(size_t)255;
  float* logits = (float*)w;    w += (((size_t)E * 4) + 255) & ~(size_t)255;
  unsigned* smax = (unsigned*)w; w += (((size_t)NSEG * 4) + 255) & ~(size_t)255;
  float* den = (float*)w;

  const int nTiles = (E + 15) / 16;

  k_init_seg<<<(NSEG + 255) / 256, 256, 0, stream>>>(smax, den, NSEG);
  k_bilinear_M<<<256, 256, 0, stream>>>(Wq, Wk, Mh);
  k_logits_wmma<<<(nTiles + NWAVE - 1) / NWAVE, 256, 0, stream>>>(
      memb, rel, qsrc, qrel, Mh, src_idx, dst_idx, query_idx, seg_ids,
      logits, smax, E, nTiles);
  k_seg_exp<<<(E + 255) / 256, 256, 0, stream>>>(logits, seg_ids, smax, den, out, E);
  k_seg_div<<<(E + 255) / 256, 256, 0, stream>>>(seg_ids, den, out, E);
}
